// Poly_Kernel_Inception_Block_69561290326526
// MI455X (gfx1250) — matrix operand order repaired and checked by reading, not compiled, not run on hardware
//
#include <hip/hip_runtime.h>
#include <hip/hip_bf16.h>

typedef __attribute__((ext_vector_type(16))) __bf16 v16bf;
typedef __attribute__((ext_vector_type(8)))  float  v8f;
typedef __attribute__((ext_vector_type(4)))  unsigned int v4u;
typedef __attribute__((ext_vector_type(8)))  int    v8i;
typedef __attribute__((ext_vector_type(4)))  int    v4i;

#define EPS_BN 1e-5f
#define CCH 256      // channels (in == hid == out)
#define HW  4096     // 64*64 pixels per image
#define NB  16       // batch
#define SLAB_STRIDE 129   // 128 cols + 1 dword TDM pad (bank-conflict-free)

// ---------------------------------------------------------------------------
// Pack 1x1-conv weights (fp32, BN-scale folded) into bf16 WMMA A-fragment
// layout.  A-matrix 16x32 bf16 (M x K) per ISA 7.12.2:
//   lanes 0-15  : row M=lane,    K = {0..7} in elems 0..7, {16..23} in 8..15
//   lanes 16-31 : row M=lane-16, K = {8..15} in elems 0..7, {24..31} in 8..15
// Tile (mt,kt) stored contiguously: wpack[((mt*8+kt)*32 + lane)*16 + e]
// Also emits BN shift vector (beta - mean*gamma*rsqrt(var+eps)).
// ---------------------------------------------------------------------------
__global__ void pack_w_kernel(const float* __restrict__ w,
                              const float* __restrict__ g,
                              const float* __restrict__ be,
                              const float* __restrict__ mu,
                              const float* __restrict__ var,
                              __bf16* __restrict__ wpack,
                              float* __restrict__ shift) {
  int idx  = blockIdx.x * blockDim.x + threadIdx.x;   // 0..65535
  int e    = idx & 15;
  int lane = (idx >> 4) & 31;
  int kt   = (idx >> 9) & 7;
  int mt   = idx >> 12;
  int m    = mt * 16 + (lane & 15);
  int kOff = ((lane >> 4) & 1) * 8 + (e & 7) + (e >> 3) * 16;
  int k    = kt * 32 + kOff;
  float sc = g[m] * rsqrtf(var[m] + EPS_BN);
  wpack[idx] = (__bf16)(w[m * CCH + k] * sc);
  if (idx < CCH)
    shift[idx] = be[idx] - mu[idx] * g[idx] * rsqrtf(var[idx] + EPS_BN);
}

// ---------------------------------------------------------------------------
// Collapse identity + dw3/dw5/dw7/dw9 into one 9x9 weight per channel,
// and sum the four per-channel biases.
// ---------------------------------------------------------------------------
__global__ void dw_prep_kernel(const float* __restrict__ w3, const float* __restrict__ b3,
                               const float* __restrict__ w5, const float* __restrict__ b5,
                               const float* __restrict__ w7, const float* __restrict__ b7,
                               const float* __restrict__ w9, const float* __restrict__ b9,
                               float* __restrict__ W9, float* __restrict__ bsum) {
  int c = blockIdx.x;
  int t = threadIdx.x;            // 96 threads
  if (t < 81) {
    int dy = t / 9 - 4, dx = t % 9 - 4;
    int ay = dy < 0 ? -dy : dy, ax = dx < 0 ? -dx : dx;
    float acc = (dy == 0 && dx == 0) ? 1.0f : 0.0f;        // identity branch
    if (ay <= 1 && ax <= 1) acc += w3[c * 9  + (dy + 1) * 3 + (dx + 1)];
    if (ay <= 2 && ax <= 2) acc += w5[c * 25 + (dy + 2) * 5 + (dx + 2)];
    if (ay <= 3 && ax <= 3) acc += w7[c * 49 + (dy + 3) * 7 + (dx + 3)];
    acc += w9[c * 81 + (dy + 4) * 9 + (dx + 4)];
    W9[c * 81 + t] = acc;
  }
  if (t == 81) bsum[c] = b3[c] + b5[c] + b7[c] + b9[c];
}

// ---------------------------------------------------------------------------
// Fused 1x1 conv + BN + ReLU as bf16 WMMA GEMM, TDM-staged.
// Block = 8 waves; block tile = 128 pixels (8 n-tiles) x 64 out-ch (4 m-tiles
// per wave).  Per K-step the Tensor Data Mover copies the 32x128 fp32
// activation slab into LDS (pad_interval=128dw/amount=1dw -> row stride 129,
// bank-conflict-free for the half-wave-split B fragment reads).  One wave
// issues the TDM descriptor, waits on TENSORcnt, then the block syncs.
// BN shift preloaded into the accumulators; ReLU fused into the store.
// ---------------------------------------------------------------------------
__global__ __launch_bounds__(256) void gemm16_kernel(
    const float*  __restrict__ in,     // [b][k][p] fp32
    const __bf16* __restrict__ wpack,  // packed A fragments
    const float*  __restrict__ shift,  // per-out-channel BN shift
    float*        __restrict__ out) {  // [b][m][p] fp32
  __shared__ float slab[32 * SLAB_STRIDE];     // 16512 B
  const int lane   = threadIdx.x & 31;
  const int wave   = threadIdx.x >> 5;
  const int lane16 = lane & 15;
  const int hi     = (lane >> 4) & 1;
  const int mg     = blockIdx.y;               // m-group: 4 m-tiles
  const int b      = blockIdx.x >> 5;          // 32 n-blocks per image
  const int p0blk  = (blockIdx.x & 31) * 128;  // block pixel base
  const int col    = wave * 16 + lane16;       // this lane's slab column

  v8f acc0, acc1, acc2, acc3;
#pragma unroll
  for (int r = 0; r < 8; ++r) {
    acc0[r] = shift[(mg * 4 + 0) * 16 + hi * 8 + r];
    acc1[r] = shift[(mg * 4 + 1) * 16 + hi * 8 + r];
    acc2[r] = shift[(mg * 4 + 2) * 16 + hi * 8 + r];
    acc3[r] = shift[(mg * 4 + 3) * 16 + hi * 8 + r];
  }

  const unsigned long long gbase =
      (unsigned long long)(const void*)(in + (size_t)b * (CCH * HW) + p0blk);
  const unsigned int lds_off = (unsigned int)(unsigned long long)(const void*)slab;
  const __bf16* wp = wpack + lane * 16;

  for (int kt = 0; kt < 8; ++kt) {
    __syncthreads();                           // readers done with slab
    if (wave == 0) {
      unsigned long long gaddr = gbase + (unsigned long long)kt * (32ull * HW * 4ull);
      v4u g0;
      g0[0] = 1u;                              // count=1 (valid descriptor)
      g0[1] = lds_off;                         // LDS byte address of slab
      g0[2] = (unsigned int)gaddr;             // global_addr[31:0]
      g0[3] = (unsigned int)((gaddr >> 32) & 0x01FFFFFFull) | 0x80000000u; // type=2
      v8i g1;
      g1[0] = 0x00020000 | (1 << 20) | (6 << 22); // data_size=4B, pad_en, 128dw interval, 1dw pad
      g1[1] = (HW & 0xFFFF) << 16;             // tensor_dim0 = 4096 (lo16)
      g1[2] = ((HW >> 16) & 0xFFFF) | ((8192 & 0xFFFF) << 16); // dim0 hi | dim1 lo (8192)
      g1[3] = ((8192 >> 16) & 0xFFFF) | (128 << 16);           // dim1 hi | tile_dim0=128
      g1[4] = 32;                              // tile_dim1 = 32 K-rows
      g1[5] = HW;                              // tensor_dim0_stride = 4096 (lo32)
      g1[6] = 0;
      g1[7] = 0;
      v4i gz4 = {0, 0, 0, 0};                  // 2D tensor: groups 2/3 disabled
      v8i gz8 = {0, 0, 0, 0, 0, 0, 0, 0};
      __builtin_amdgcn_tensor_load_to_lds(g0, g1, gz4, gz4, gz8, 0);
      __builtin_amdgcn_s_wait_tensorcnt(0);
    }
    __syncthreads();                           // slab visible to all waves

    // B fragment from LDS: elem e -> K = (e&7) + 8*hi + 16*(e>>3), column fixed per lane.
    v16bf bf;
#pragma unroll
    for (int e = 0; e < 16; ++e)
      bf[e] = (__bf16)slab[((e & 7) + hi * 8 + (e >> 3) * 16) * SLAB_STRIDE + col];

    // 4 m-tiles share one B fragment.
    const __bf16* wk = wp + ((mg * 4 * 8 + kt) * 32) * 16;
    v16bf a0 = *(const v16bf*)(wk + 0 * 8 * 32 * 16);
    v16bf a1 = *(const v16bf*)(wk + 1 * 8 * 32 * 16);
    v16bf a2 = *(const v16bf*)(wk + 2 * 8 * 32 * 16);
    v16bf a3 = *(const v16bf*)(wk + 3 * 8 * 32 * 16);
    acc0 = __builtin_amdgcn_wmma_f32_16x16x32_bf16(false, a0, false, bf, (short)0, acc0, false, false);
    acc1 = __builtin_amdgcn_wmma_f32_16x16x32_bf16(false, a1, false, bf, (short)0, acc1, false, false);
    acc2 = __builtin_amdgcn_wmma_f32_16x16x32_bf16(false, a2, false, bf, (short)0, acc2, false, false);
    acc3 = __builtin_amdgcn_wmma_f32_16x16x32_bf16(false, a3, false, bf, (short)0, acc3, false, false);
  }

  float* outb = out + (size_t)b * (CCH * HW) + p0blk + col;
#pragma unroll
  for (int r = 0; r < 8; ++r) {
    float v0 = acc0[r], v1 = acc1[r], v2 = acc2[r], v3 = acc3[r];
    outb[(size_t)((mg * 4 + 0) * 16 + hi * 8 + r) * HW] = v0 > 0.0f ? v0 : 0.0f;
    outb[(size_t)((mg * 4 + 1) * 16 + hi * 8 + r) * HW] = v1 > 0.0f ? v1 : 0.0f;
    outb[(size_t)((mg * 4 + 2) * 16 + hi * 8 + r) * HW] = v2 > 0.0f ? v2 : 0.0f;
    outb[(size_t)((mg * 4 + 3) * 16 + hi * 8 + r) * HW] = v3 > 0.0f ? v3 : 0.0f;
  }
}

// ---------------------------------------------------------------------------
// s = h + sum_k (dw_k(h) + b_k) as one 9x9 stencil from an LDS halo tile.
// Block = 32x32 output pixels (4 per thread); halo 4 -> 40x40 tile (stride 41).
// ---------------------------------------------------------------------------
__global__ __launch_bounds__(256) void dw_sum_kernel(
    const float* __restrict__ h, const float* __restrict__ W9,
    const float* __restrict__ bsum, float* __restrict__ s) {
  __shared__ float tile[40 * 41];
  __shared__ float wsh[81];
  const int c   = blockIdx.y;
  const int b   = blockIdx.z;
  const int th  = (blockIdx.x >> 1) * 32;
  const int tw  = (blockIdx.x & 1) * 32;
  const int tid = threadIdx.x;
  const float* hp = h + ((size_t)b * CCH + c) * HW;

  if (tid < 81) wsh[tid] = W9[c * 81 + tid];
  for (int i = tid; i < 40 * 40; i += 256) {
    int ly = i / 40, lx = i % 40;
    int gy = th + ly - 4, gx = tw + lx - 4;
    float v = 0.0f;
    if (gy >= 0 && gy < 64 && gx >= 0 && gx < 64) v = hp[gy * 64 + gx];
    tile[ly * 41 + lx] = v;
  }
  __syncthreads();

  const int tx = tid & 31, ty0 = tid >> 5;     // rows ty0, ty0+8, +16, +24
  float a0 = bsum[c], a1 = a0, a2 = a0, a3 = a0;
  for (int dy = 0; dy < 9; ++dy) {
#pragma unroll
    for (int dx = 0; dx < 9; ++dx) {
      float w = wsh[dy * 9 + dx];
      a0 += tile[(ty0 +  0 + dy) * 41 + tx + dx] * w;
      a1 += tile[(ty0 +  8 + dy) * 41 + tx + dx] * w;
      a2 += tile[(ty0 + 16 + dy) * 41 + tx + dx] * w;
      a3 += tile[(ty0 + 24 + dy) * 41 + tx + dx] * w;
    }
  }
  float* sp = s + ((size_t)b * CCH + c) * HW;
  sp[(th + ty0 +  0) * 64 + tw + tx] = a0;
  sp[(th + ty0 +  8) * 64 + tw + tx] = a1;
  sp[(th + ty0 + 16) * 64 + tw + tx] = a2;
  sp[(th + ty0 + 24) * 64 + tw + tx] = a3;
}

// ---------------------------------------------------------------------------
extern "C" void kernel_launch(void* const* d_in, const int* in_sizes, int n_in,
                              void* d_out, int out_size, void* d_ws, size_t ws_size,
                              hipStream_t stream) {
  const float* x      = (const float*)d_in[0];
  const float* pre_w  = (const float*)d_in[1];
  const float* bn1_g  = (const float*)d_in[2];
  const float* bn1_b  = (const float*)d_in[3];
  const float* bn1_m  = (const float*)d_in[4];
  const float* bn1_v  = (const float*)d_in[5];
  const float* dw3_w  = (const float*)d_in[6];
  const float* dw3_b  = (const float*)d_in[7];
  const float* dw5_w  = (const float*)d_in[8];
  const float* dw5_b  = (const float*)d_in[9];
  const float* dw7_w  = (const float*)d_in[10];
  const float* dw7_b  = (const float*)d_in[11];
  const float* dw9_w  = (const float*)d_in[12];
  const float* dw9_b  = (const float*)d_in[13];
  const float* pw_w   = (const float*)d_in[14];
  const float* bn2_g  = (const float*)d_in[15];
  const float* bn2_b  = (const float*)d_in[16];
  const float* bn2_m  = (const float*)d_in[17];
  const float* bn2_v  = (const float*)d_in[18];
  const float* post_w = (const float*)d_in[19];
  const float* bn3_g  = (const float*)d_in[20];
  const float* bn3_b  = (const float*)d_in[21];
  const float* bn3_m  = (const float*)d_in[22];
  const float* bn3_v  = (const float*)d_in[23];

  char* ws = (char*)d_ws;
  __bf16* wpPre  = (__bf16*)(ws + 0);            // 128 KiB each
  __bf16* wpPw   = (__bf16*)(ws + 131072);
  __bf16* wpPost = (__bf16*)(ws + 262144);
  float*  sh1    = (float*)(ws + 393216);
  float*  sh2    = (float*)(ws + 394240);
  float*  sh3    = (float*)(ws + 395264);
  float*  W9     = (float*)(ws + 396288);        // 256*81*4 B
  float*  bsum   = (float*)(ws + 479232);
  float*  hbuf   = (float*)(ws + (size_t)(1u << 20));               // 64 MiB
  float*  sbuf   = (float*)(ws + (size_t)(1u << 20) + 67108864u);   // 64 MiB
  float*  outF   = (float*)d_out;

  // Weight prep (tiny)
  pack_w_kernel<<<256, 256, 0, stream>>>(pre_w,  bn1_g, bn1_b, bn1_m, bn1_v, wpPre,  sh1);
  pack_w_kernel<<<256, 256, 0, stream>>>(pw_w,   bn2_g, bn2_b, bn2_m, bn2_v, wpPw,   sh2);
  pack_w_kernel<<<256, 256, 0, stream>>>(post_w, bn3_g, bn3_b, bn3_m, bn3_v, wpPost, sh3);
  dw_prep_kernel<<<256, 96, 0, stream>>>(dw3_w, dw3_b, dw5_w, dw5_b,
                                         dw7_w, dw7_b, dw9_w, dw9_b, W9, bsum);

  dim3 gGemm(512, 4, 1);             // 512 n-blocks(128 px) x 4 m-groups(64 ch)
  dim3 gDw(4, CCH, NB);              // 2x2 tiles x channel x batch

  // pre_conv: x -> h
  gemm16_kernel<<<gGemm, 256, 0, stream>>>(x, wpPre, sh1, hbuf);
  // identity + 4 depthwise branches: h -> s
  dw_sum_kernel<<<gDw, 256, 0, stream>>>(hbuf, W9, bsum, sbuf);
  // pw_conv: s -> t (reuse hbuf)
  gemm16_kernel<<<gGemm, 256, 0, stream>>>(sbuf, wpPw, sh2, hbuf);
  // post_conv: t -> out
  gemm16_kernel<<<gGemm, 256, 0, stream>>>(hbuf, wpPost, sh3, outF);
}
